// GNNGroupTracker_4793183502742
// MI455X (gfx1250) — matrix operand order repaired and checked by reading, not compiled, not run on hardware
//
#include <hip/hip_runtime.h>
#include <math.h>

typedef __attribute__((ext_vector_type(16))) _Float16 v16h;
typedef __attribute__((ext_vector_type(8)))  _Float16 v8h;
typedef __attribute__((ext_vector_type(8)))  float    v8f;

#define HID 96
#define HEADS 4
#define HC 24
#define LN_EPS 1e-5f
#define ATT_SCALE 0.20412414523193154f  /* 1/sqrt(24) */

__device__ __forceinline__ float gelu_exact(float x) {
  return 0.5f * x * (1.0f + erff(x * 0.70710678118654752f));
}

// order-preserving float <-> uint encoding for atomic max
__device__ __forceinline__ unsigned f2ord(float x) {
  unsigned b = __float_as_uint(x);
  return (b & 0x80000000u) ? ~b : (b | 0x80000000u);
}
__device__ __forceinline__ float ord2f(unsigned k) {
  unsigned b = (k & 0x80000000u) ? (k ^ 0x80000000u) : ~k;
  return __uint_as_float(b);
}

// pack two 16-byte half-vectors (at p and p+16) into one WMMA fragment
__device__ __forceinline__ v16h ldfrag(const _Float16* p) {
  v8h lo = *(const v8h*)(p);
  v8h hi = *(const v8h*)(p + 16);
  v16h r;
  #pragma unroll
  for (int j = 0; j < 8; j++) { r[j] = lo[j]; r[8 + j] = hi[j]; }
  return r;
}

// ---------------------------------------------------------------------------
__global__ __launch_bounds__(256) void fill_zero_kernel(float* p, int n) {
  int i = blockIdx.x * blockDim.x + threadIdx.x;
  if (i < n) p[i] = 0.0f;
}

__global__ __launch_bounds__(256) void f32_to_f16_kernel(
    const float* __restrict__ src, _Float16* __restrict__ dst, int n) {
  int i = blockIdx.x * blockDim.x + threadIdx.x;
  if (i < n) dst[i] = (_Float16)src[i];
}

// ---------------------------------------------------------------------------
// Encoder front: t = x @ W1^T + b1 (inDim in {2,3}); LayerNorm(96); GELU.
// One wave per row, lane owns channels {lane, lane+32, lane+64}; f16 output.
__global__ __launch_bounds__(256) void enc_front_kernel(
    const float* __restrict__ x, int inDim,
    const float* __restrict__ W1, const float* __restrict__ b1,
    const float* __restrict__ g,  const float* __restrict__ be,
    _Float16* __restrict__ out, int rows) {
  int gid  = blockIdx.x * blockDim.x + threadIdx.x;
  int row  = gid >> 5;
  int lane = gid & 31;
  if (row >= rows) return;
  const float* xr = x + (size_t)row * inDim;
  float xv[3];
  #pragma unroll
  for (int i = 0; i < 3; i++) xv[i] = (i < inDim) ? xr[i] : 0.0f;
  float t[3], s = 0.0f, ss = 0.0f;
  #pragma unroll
  for (int c = 0; c < 3; c++) {
    int j = lane + 32 * c;
    const float* wr = W1 + (size_t)j * inDim;
    float acc = b1[j];
    for (int i = 0; i < inDim; i++) acc += wr[i] * xv[i];
    t[c] = acc; s += acc; ss += acc * acc;
  }
  for (int m = 16; m > 0; m >>= 1) { s += __shfl_xor(s, m, 32); ss += __shfl_xor(ss, m, 32); }
  float mean = s * (1.0f / 96.0f);
  float var  = ss * (1.0f / 96.0f) - mean * mean;
  float rstd = rsqrtf(var + LN_EPS);
  _Float16* orow = out + (size_t)row * 96;
  #pragma unroll
  for (int c = 0; c < 3; c++) {
    int j = lane + 32 * c;
    float v = (t[c] - mean) * rstd * g[j] + be[j];
    orow[j] = (_Float16)gelu_exact(v);
  }
}

// ---------------------------------------------------------------------------
// LayerNorm + GELU over D columns (D multiple of 32, D<=256); f32 in, f16 out.
__global__ __launch_bounds__(256) void ln_gelu_half_kernel(
    const float* __restrict__ X, const float* __restrict__ g,
    const float* __restrict__ be, _Float16* __restrict__ Y, int rows, int D) {
  int gid  = blockIdx.x * blockDim.x + threadIdx.x;
  int row  = gid >> 5;
  int lane = gid & 31;
  if (row >= rows) return;
  int C = D >> 5;
  float t[8], s = 0.0f, ss = 0.0f;
  const float* xr = X + (size_t)row * D;
  for (int c = 0; c < C; c++) {
    int j = lane + 32 * c;
    float v = xr[j];
    t[c] = v; s += v; ss += v * v;
  }
  for (int m = 16; m > 0; m >>= 1) { s += __shfl_xor(s, m, 32); ss += __shfl_xor(ss, m, 32); }
  float mean = s / (float)D;
  float var  = ss / (float)D - mean * mean;
  float rstd = rsqrtf(var + LN_EPS);
  _Float16* yr = Y + (size_t)row * D;
  for (int c = 0; c < C; c++) {
    int j = lane + 32 * c;
    float v = (t[c] - mean) * rstd * g[j] + be[j];
    yr[j] = (_Float16)gelu_exact(v);
  }
}

// ---------------------------------------------------------------------------
// WMMA GEMM: C[M,N] = act(A[M,K](f16,lda) @ W16[N,K](f16,ldw,+wOff)^T + bias)
// One wave per 32x32 output tile: 2 A-fragments x 2 B-fragments -> 4 WMMAs
// per K-step on 8 b128 loads (no conversions: both operands pre-f16).
// K is a template constant -> fully unrolled; clamped indices keep EXEC=all-1s.
template <int K>
__global__ __launch_bounds__(256) void gemm_wmma_kernel(
    const _Float16* __restrict__ A, int lda,
    const _Float16* __restrict__ W, int ldw, int wOff,
    const float* __restrict__ bias,
    void* __restrict__ Cout, int ldc,
    int M, int N, int act, int outHalf) {
  int wid  = (blockIdx.x * blockDim.x + threadIdx.x) >> 5;
  int lane = threadIdx.x & 31;
  int mtiles = (M + 31) >> 5;
  int ntiles = (N + 31) >> 5;
  if (wid >= mtiles * ntiles) return;       // wave-uniform exit
  int mt = wid / ntiles;
  int nt = wid - mt * ntiles;
  int m0 = mt << 5, n0 = nt << 5;
  int l15 = lane & 15;
  int hi  = lane >> 4;                      // 0: k 0-7,16-23; 1: k 8-15,24-31
  int r0 = m0 + l15;      if (r0 > M - 1) r0 = M - 1;
  int r1 = m0 + 16 + l15; if (r1 > M - 1) r1 = M - 1;
  int c0 = n0 + l15;      if (c0 > N - 1) c0 = N - 1;
  int c1 = n0 + 16 + l15; if (c1 > N - 1) c1 = N - 1;
  int ko = hi ? 8 : 0;
  const _Float16* a0p = A + (size_t)r0 * lda + ko;
  const _Float16* a1p = A + (size_t)r1 * lda + ko;
  const _Float16* b0p = W + (size_t)c0 * ldw + wOff + (hi ? 8 : 0);
  const _Float16* b1p = W + (size_t)c1 * ldw + wOff + (hi ? 8 : 0);
  v8f acc00 = {}, acc01 = {}, acc10 = {}, acc11 = {};
  #pragma unroll
  for (int kb = 0; kb < K; kb += 32) {
    v16h a0 = ldfrag(a0p + kb);
    v16h a1 = ldfrag(a1p + kb);
    v16h b0 = ldfrag(b0p + kb);
    v16h b1 = ldfrag(b1p + kb);
    acc00 = __builtin_amdgcn_wmma_f32_16x16x32_f16(false, a0, false, b0, (short)0, acc00, false, false);
    acc01 = __builtin_amdgcn_wmma_f32_16x16x32_f16(false, a0, false, b1, (short)0, acc01, false, false);
    acc10 = __builtin_amdgcn_wmma_f32_16x16x32_f16(false, a1, false, b0, (short)0, acc10, false, false);
    acc11 = __builtin_amdgcn_wmma_f32_16x16x32_f16(false, a1, false, b1, (short)0, acc11, false, false);
  }
  #pragma unroll
  for (int rt = 0; rt < 2; rt++) {
    int rbase = m0 + rt * 16 + (hi ? 8 : 0);
    #pragma unroll
    for (int ct = 0; ct < 2; ct++) {
      int ccol = n0 + ct * 16 + l15;
      if (ccol < N) {
        float bv = bias ? bias[ccol] : 0.0f;
        #pragma unroll
        for (int i = 0; i < 8; i++) {
          int r = rbase + i;
          if (r < M) {
            float v = (rt ? (ct ? acc11[i] : acc10[i])
                          : (ct ? acc01[i] : acc00[i])) + bv;
            if (act) v = gelu_exact(v);
            if (outHalf) ((_Float16*)Cout)[(size_t)r * ldc + ccol] = (_Float16)v;
            else         ((float*)Cout)[(size_t)r * ldc + ccol] = v;
          }
        }
      }
    }
  }
}

// ---------------------------------------------------------------------------
// Attention: alpha[e,h] = SCALE * sum_c q[dst][h,c]*(k[src][h,c]+ee[e][h,c])
__global__ __launch_bounds__(256) void edge_alpha_kernel(
    const _Float16* __restrict__ q, const _Float16* __restrict__ k,
    const _Float16* __restrict__ ee,
    const int* __restrict__ src, const int* __restrict__ dst,
    float* __restrict__ alpha, unsigned* __restrict__ amax, int E) {
  int tid = blockIdx.x * blockDim.x + threadIdx.x;
  if (tid >= E * HEADS) return;
  int eid = tid >> 2, hh = tid & 3;
  int s = src[eid], d = dst[eid];
  const _Float16* qp = q  + (size_t)d   * HID + hh * HC;
  const _Float16* kp = k  + (size_t)s   * HID + hh * HC;
  const _Float16* ep = ee + (size_t)eid * HID + hh * HC;
  float acc = 0.0f;
  #pragma unroll
  for (int u = 0; u < 3; u++) {
    v8h qv = *(const v8h*)(qp + 8 * u);
    v8h kv = *(const v8h*)(kp + 8 * u);
    v8h ev = *(const v8h*)(ep + 8 * u);
    #pragma unroll
    for (int j = 0; j < 8; j++)
      acc += (float)qv[j] * ((float)kv[j] + (float)ev[j]);
  }
  float al = acc * ATT_SCALE;
  alpha[tid] = al;
  atomicMax(&amax[(size_t)d * HEADS + hh], f2ord(al));
}

__global__ __launch_bounds__(256) void edge_exp_kernel(
    const int* __restrict__ dst, const unsigned* __restrict__ amax,
    float* __restrict__ alpha, float* __restrict__ den, int E) {
  int tid = blockIdx.x * blockDim.x + threadIdx.x;
  if (tid >= E * HEADS) return;
  int eid = tid >> 2, hh = tid & 3;
  int d = dst[eid];
  float am = ord2f(amax[(size_t)d * HEADS + hh]);
  float ex = expf(alpha[tid] - am);
  alpha[tid] = ex;
  atomicAdd(&den[(size_t)d * HEADS + hh], ex);
}

__global__ __launch_bounds__(256) void edge_msg_kernel(
    const _Float16* __restrict__ v, const _Float16* __restrict__ ee,
    const int* __restrict__ src, const int* __restrict__ dst,
    const float* __restrict__ alpha, const float* __restrict__ den,
    float* __restrict__ out, int E) {
  int tid = blockIdx.x * blockDim.x + threadIdx.x;
  if (tid >= E * HEADS) return;
  int eid = tid >> 2, hh = tid & 3;
  int s = src[eid], d = dst[eid];
  float a = alpha[tid] / den[(size_t)d * HEADS + hh];
  const _Float16* vp = v  + (size_t)s   * HID + hh * HC;
  const _Float16* ep = ee + (size_t)eid * HID + hh * HC;
  float* op = out + (size_t)d * HID + hh * HC;
  #pragma unroll
  for (int u = 0; u < 3; u++) {
    v8h vv = *(const v8h*)(vp + 8 * u);
    v8h ev = *(const v8h*)(ep + 8 * u);
    #pragma unroll
    for (int j = 0; j < 8; j++)
      atomicAdd(&op[8 * u + j], ((float)vv[j] + (float)ev[j]) * a);
  }
}

// ---------------------------------------------------------------------------
// BatchNorm over node dim: one block per channel.
__global__ __launch_bounds__(256) void bn_stats_kernel(
    const float* __restrict__ X, int rows,
    float* __restrict__ mean, float* __restrict__ rstd) {
  __shared__ float s1[256], s2[256];
  int ch = blockIdx.x, t = threadIdx.x;
  float s = 0.0f, ss = 0.0f;
  for (int n = t; n < rows; n += 256) {
    float v = X[(size_t)n * HID + ch];
    s += v; ss += v * v;
  }
  s1[t] = s; s2[t] = ss;
  __syncthreads();
  for (int o = 128; o > 0; o >>= 1) {
    if (t < o) { s1[t] += s1[t + o]; s2[t] += s2[t + o]; }
    __syncthreads();
  }
  if (t == 0) {
    float m = s1[0] / (float)rows;
    float v = s2[0] / (float)rows - m * m;
    mean[ch] = m;
    rstd[ch] = rsqrtf(v + LN_EPS);
  }
}

// hc = gelu(bn(x)) + hprev ; writes f16 column slice of hcat16 (ldo=384)
__global__ __launch_bounds__(256) void bn_apply_kernel(
    const float* __restrict__ X, const float* __restrict__ mean,
    const float* __restrict__ rstd, const float* __restrict__ g,
    const float* __restrict__ b,
    const _Float16* __restrict__ hprev, int ldp,
    _Float16* __restrict__ hout, int ldo, int rows) {
  int idx = blockIdx.x * blockDim.x + threadIdx.x;
  if (idx >= rows * HID) return;
  int n = idx / HID, ch = idx - n * HID;
  float v = (X[idx] - mean[ch]) * rstd[ch] * g[ch] + b[ch];
  float hc = gelu_exact(v) + (float)hprev[(size_t)n * ldp + ch];
  hout[(size_t)n * ldo + ch] = (_Float16)hc;
}

// ---------------------------------------------------------------------------
// Edge classifier tail: z = P[src]+Q[dst]+R[e]+b1; score = sigmoid(gelu(z)·w2+b2)
__global__ __launch_bounds__(256) void edge_score_kernel(
    const _Float16* __restrict__ P, const _Float16* __restrict__ Q,
    const _Float16* __restrict__ R,
    const int* __restrict__ src, const int* __restrict__ dst,
    const float* __restrict__ b1, const float* __restrict__ w2,
    const float* __restrict__ b2,
    float* __restrict__ out, int E) {
  __shared__ float sb[HID], sw[HID];
  if (threadIdx.x < HID) {
    sb[threadIdx.x] = b1[threadIdx.x];
    sw[threadIdx.x] = w2[threadIdx.x];
  }
  __syncthreads();
  int eid = blockIdx.x * blockDim.x + threadIdx.x;
  if (eid >= E) return;
  int s = src[eid], d = dst[eid];
  const _Float16* Pp = P + (size_t)s   * HID;
  const _Float16* Qp = Q + (size_t)d   * HID;
  const _Float16* Rp = R + (size_t)eid * HID;
  float acc = 0.0f;
  #pragma unroll
  for (int u = 0; u < 12; u++) {
    v8h pv = *(const v8h*)(Pp + 8 * u);
    v8h qv = *(const v8h*)(Qp + 8 * u);
    v8h rv = *(const v8h*)(Rp + 8 * u);
    #pragma unroll
    for (int j = 0; j < 8; j++) {
      int c = 8 * u + j;
      acc += gelu_exact((float)pv[j] + (float)qv[j] + (float)rv[j] + sb[c]) * sw[c];
    }
  }
  float z = acc + b2[0];
  out[eid] = 1.0f / (1.0f + expf(-z));
}

// ===========================================================================
extern "C" void kernel_launch(void* const* d_in, const int* in_sizes, int n_in,
                              void* d_out, int out_size, void* d_ws, size_t ws_size,
                              hipStream_t stream) {
  (void)n_in; (void)out_size; (void)ws_size;
  const float* x  = (const float*)d_in[0];
  const int*   ei = (const int*)  d_in[1];
  const float* ea = (const float*)d_in[2];
  const int N = in_sizes[0] / 2;       // 50000
  const int E = in_sizes[2] / 3;       // 400000
  const int* src = ei;
  const int* dst = ei + E;

  enum {
    NE_W1 = 3, NE_B1, NE_G, NE_BE, NE_W2, NE_B2,
    EE_W1, EE_B1, EE_G, EE_BE, EE_W2, EE_B2,
    FU_W1, FU_B1, FU_G, FU_BE, FU_W2, FU_B2,
    EC_W1, EC_B1, EC_W2, EC_B2,
    OR_W1, OR_B1, OR_W2, OR_B2,
    LBASE = 29  // + layer*11 : Wq,bq,Wk,bk,Wv,bv,We,Ws,bs,bn_g,bn_b
  };
  auto pp = [&](int i) { return (const float*)d_in[i]; };

  // ---- workspace carve (256B-aligned blocks) ------------------------------
  char* base = (char*)d_ws;
  size_t off = 0;
  auto takeB = [&](size_t bytes) {
    void* r = base + off;
    off += (bytes + 255) & ~(size_t)255;
    return r;
  };
  _Float16* e16   = (_Float16*)takeB((size_t)E * HID * 2);   // encoded edges (persistent)
  _Float16* ee16  = (_Float16*)takeB((size_t)E * HID * 2);   // e@We per layer; later R; edge-front temp
  float*    alpha = (float*)   takeB((size_t)E * HEADS * 4);
  float*    astat = (float*)   takeB((size_t)N * HEADS * 8); // [amax(u32) | den(f32)]
  unsigned* amaxu = (unsigned*)astat;
  float*    den   = astat + (size_t)N * HEADS;
  float*    tmp   = (float*)   takeB((size_t)N * HID * 4);   // skip + scattered msgs (atomics)
  float*    hfu   = (float*)   takeB((size_t)N * 2 * HID * 4);
  _Float16* h016  = (_Float16*)takeB((size_t)N * HID * 2);
  _Float16* q16   = (_Float16*)takeB((size_t)N * HID * 2);   // q; later P; node-front temp
  _Float16* k16   = (_Float16*)takeB((size_t)N * HID * 2);   // k; later Q2
  _Float16* v16   = (_Float16*)takeB((size_t)N * HID * 2);   // v; later offset hidden
  _Float16* hcat16= (_Float16*)takeB((size_t)N * 4 * HID * 2);
  _Float16* hfu16 = (_Float16*)takeB((size_t)N * 2 * HID * 2);
  _Float16* hf16  = (_Float16*)takeB((size_t)N * HID * 2);
  float*    bnm   = (float*)   takeB(HID * 4);
  float*    bnr   = (float*)   takeB(HID * 4);

  float* out_scores  = (float*)d_out;   // [E]
  float* out_offsets = out_scores + E;  // [N,2]

  auto cdiv = [](long a, long b) { return (int)((a + b - 1) / b); };

  // ---- pre-convert all GEMM weight matrices to f16 (stay hot in L2) ------
  auto cvtW = [&](int idx, size_t n) {
    _Float16* dstp = (_Float16*)takeB(n * 2);
    f32_to_f16_kernel<<<cdiv((long)n, 256), 256, 0, stream>>>(pp(idx), dstp, (int)n);
    return dstp;
  };
  _Float16* ne_W2h = cvtW(NE_W2, 96 * 96);
  _Float16* ee_W2h = cvtW(EE_W2, 96 * 96);
  _Float16* fu_W1h = cvtW(FU_W1, 192 * 384);
  _Float16* fu_W2h = cvtW(FU_W2, 96 * 192);
  _Float16* ec_W1h = cvtW(EC_W1, 96 * 288);
  _Float16* or_W1h = cvtW(OR_W1, 96 * 96);
  _Float16* or_W2h = cvtW(OR_W2, 2 * 96);
  _Float16* Wqh[4], *Wkh[4], *Wvh[4], *Weh[4], *Wsh[4];
  for (int L = 0; L < 4; L++) {
    int B = LBASE + L * 11;
    Wqh[L] = cvtW(B + 0, 96 * 96);
    Wkh[L] = cvtW(B + 2, 96 * 96);
    Wvh[L] = cvtW(B + 4, 96 * 96);
    Weh[L] = cvtW(B + 6, 96 * 96);
    Wsh[L] = cvtW(B + 7, 96 * 96);
  }

  auto gemm = [&](const _Float16* A, int lda, const _Float16* W, int ldw, int wOff,
                  const float* bias, void* C, int ldc,
                  long M, int Nc, int K, int act, int outHalf) {
    long waves  = ((M + 31) / 32) * ((Nc + 31) / 32);
    int  blocks = (int)((waves * 32 + 255) / 256);
    if (K == 96)
      gemm_wmma_kernel<96><<<blocks, 256, 0, stream>>>(A, lda, W, ldw, wOff, bias, C, ldc, (int)M, Nc, act, outHalf);
    else if (K == 192)
      gemm_wmma_kernel<192><<<blocks, 256, 0, stream>>>(A, lda, W, ldw, wOff, bias, C, ldc, (int)M, Nc, act, outHalf);
    else
      gemm_wmma_kernel<384><<<blocks, 256, 0, stream>>>(A, lda, W, ldw, wOff, bias, C, ldc, (int)M, Nc, act, outHalf);
  };

  // ---- node / edge encoders ----------------------------------------------
  enc_front_kernel<<<cdiv((long)N * 32, 256), 256, 0, stream>>>(
      x, 2, pp(NE_W1), pp(NE_B1), pp(NE_G), pp(NE_BE), q16, N);
  gemm(q16, HID, ne_W2h, HID, 0, pp(NE_B2), h016, HID, N, HID, HID, 0, 1);

  enc_front_kernel<<<cdiv((long)E * 32, 256), 256, 0, stream>>>(
      ea, 3, pp(EE_W1), pp(EE_B1), pp(EE_G), pp(EE_BE), ee16, E);
  gemm(ee16, HID, ee_W2h, HID, 0, pp(EE_B2), e16, HID, E, HID, HID, 0, 1);

  // ---- 4 TransformerConv layers ------------------------------------------
  const _Float16* h = h016;
  int ldh = HID;
  for (int L = 0; L < 4; L++) {
    int B = LBASE + L * 11;
    gemm(h, ldh, Wqh[L], HID, 0, pp(B + 1), q16, HID, N, HID, HID, 0, 1);  // q
    gemm(h, ldh, Wkh[L], HID, 0, pp(B + 3), k16, HID, N, HID, HID, 0, 1);  // k
    gemm(h, ldh, Wvh[L], HID, 0, pp(B + 5), v16, HID, N, HID, HID, 0, 1);  // v
    gemm(e16, HID, Weh[L], HID, 0, nullptr, ee16, HID, E, HID, HID, 0, 1); // ee
    gemm(h, ldh, Wsh[L], HID, 0, pp(B + 8), tmp, HID, N, HID, HID, 0, 0);  // root skip (f32)

    fill_zero_kernel<<<cdiv((long)N * 2 * HEADS, 256), 256, 0, stream>>>(
        astat, N * 2 * HEADS);  // amax (encoded -inf) and den

    int eThreads = E * HEADS;
    edge_alpha_kernel<<<cdiv(eThreads, 256), 256, 0, stream>>>(
        q16, k16, ee16, src, dst, alpha, amaxu, E);
    edge_exp_kernel<<<cdiv(eThreads, 256), 256, 0, stream>>>(
        dst, amaxu, alpha, den, E);
    edge_msg_kernel<<<cdiv(eThreads, 256), 256, 0, stream>>>(
        v16, ee16, src, dst, alpha, den, tmp, E);

    bn_stats_kernel<<<HID, 256, 0, stream>>>(tmp, N, bnm, bnr);
    bn_apply_kernel<<<cdiv((long)N * HID, 256), 256, 0, stream>>>(
        tmp, bnm, bnr, pp(B + 9), pp(B + 10), h, ldh, hcat16 + L * HID, 4 * HID, N);

    h = hcat16 + L * HID;
    ldh = 4 * HID;
  }

  // ---- fusion MLP ---------------------------------------------------------
  gemm(hcat16, 4 * HID, fu_W1h, 4 * HID, 0, pp(FU_B1), hfu, 2 * HID,
       N, 2 * HID, 4 * HID, 0, 0);
  ln_gelu_half_kernel<<<cdiv((long)N * 32, 256), 256, 0, stream>>>(
      hfu, pp(FU_G), pp(FU_BE), hfu16, N, 2 * HID);
  gemm(hfu16, 2 * HID, fu_W2h, 2 * HID, 0, pp(FU_B2), hf16, HID, N, HID, 2 * HID, 0, 1);

  // ---- edge classifier: ec_W1 [96,288] split into three 96x96 blocks ------
  gemm(hf16, HID, ec_W1h, 3 * HID, 0,       nullptr, q16,  HID, N, HID, HID, 0, 1); // P (src)
  gemm(hf16, HID, ec_W1h, 3 * HID, HID,     nullptr, k16,  HID, N, HID, HID, 0, 1); // Q (dst)
  gemm(e16,  HID, ec_W1h, 3 * HID, 2 * HID, nullptr, ee16, HID, E, HID, HID, 0, 1); // R (edge)
  edge_score_kernel<<<cdiv((long)E, 256), 256, 0, stream>>>(
      q16, k16, ee16, src, dst, pp(EC_B1), pp(EC_W2), pp(EC_B2), out_scores, E);

  // ---- offset regressor ---------------------------------------------------
  gemm(hf16, HID, or_W1h, HID, 0, pp(OR_B1), v16, HID, N, HID, HID, 1, 1);  // GELU fused
  gemm(v16,  HID, or_W2h, HID, 0, pp(OR_B2), out_offsets, 2, N, 2, HID, 0, 0);
}
